// MultiHeadAttentionLayerRPE_77781857731287
// MI455X (gfx1250) — matrix operand order repaired and checked by reading, not compiled, not run on hardware
//
#include <hip/hip_runtime.h>

typedef __attribute__((ext_vector_type(16))) _Float16 v16h;
typedef __attribute__((ext_vector_type(8)))  _Float16 h8;
typedef __attribute__((ext_vector_type(8)))  float    v8f;

#define D_MODEL 1024
#define NUM_HEADS 16
#define DK 64
#define SEQ 2048
#define PMAX 16
#define NREL 33   // 2*PMAX+1

// ---------------------------------------------------------------------------
// WMMA helpers (CDNA5 v_wmma_f32_16x16x32_f16, wave32)
// ---------------------------------------------------------------------------
__device__ __forceinline__ v8f wmma16(v16h a, v16h b, v8f c) {
  // (neg_a, A, neg_b, B, c_mod, C, reuse_a, reuse_b)
  return __builtin_amdgcn_wmma_f32_16x16x32_f16(false, a, false, b, (short)0, c,
                                                false, false);
}

// A fragment: 16x32 f16 tile, row-major in LDS with leading dim `ld` (halves).
// Layout: lane L -> m = L&15, g = L>>4; VGPR0..3 hold K = g*16 + 0..7,
// VGPR4..7 hold K = g*16 + 8..15  => two contiguous 16B loads per lane.
__device__ __forceinline__ v16h frag_a(const _Float16* base, int ld) {
  const int lane = threadIdx.x & 31;
  const int m = lane & 15, g = lane >> 4;
  const _Float16* r = base + m * ld;
  h8 lo = *(const h8*)(r + g * 16);
  h8 hi = *(const h8*)(r + g * 16 + 8);
  v16h a;
#pragma unroll
  for (int i = 0; i < 8; ++i) { a[i] = lo[i]; a[i + 8] = hi[i]; }
  return a;
}

// B fragment: 32x16 f16 (KxN) loaded from a row-major B^T tile (rows = N,
// cols = K, leading dim `ld`). Layout (same K order as frag_a): lane L -> n = L&15, g = L>>4 holds
// K = g*16 .. g*16+15 => one contiguous 32B load per lane.
__device__ __forceinline__ v16h frag_b(const _Float16* baseT, int ld) {
  const int lane = threadIdx.x & 31;
  const int n = lane & 15, g = lane >> 4;
  return *(const v16h*)(baseT + n * ld + g * 16);
}

// ---------------------------------------------------------------------------
// GEMM: Y[M=2048][N=1024] = A[2048][1024] @ W^T + bias   (W is [N][K] row-major)
// mode 0: Y -> f16 head-major  Out[h][s][64]      (Q, K projections)
// mode 1: Y -> f16 transposed  Out[h][64][s]      (V projection, stores V^T)
// mode 2: Y -> f32 plain       Out[s][1024]       (final output projection)
// Block: 256 thr (8 waves), tile 64(M) x 128(N), wave tile 32x32, K-step 64.
// ---------------------------------------------------------------------------
__global__ __launch_bounds__(256) void gemm_xwt_kernel(
    const void* __restrict__ Ain, int a_f16,
    const float* __restrict__ W, const float* __restrict__ bias,
    void* __restrict__ Out, int mode)
{
  __shared__ __align__(32) _Float16 As[64 * 64];
  __shared__ __align__(32) _Float16 Bs[128 * 64];

  const int tid = threadIdx.x;
  const int m0 = blockIdx.y * 64;
  const int n0 = blockIdx.x * 128;
  const int wave = tid >> 5, lane = tid & 31;
  const int wm = wave & 1, wn = wave >> 1;

  v8f acc[2][2] = {};

  for (int k0 = 0; k0 < D_MODEL; k0 += 64) {
    {  // stage A tile (64x64 halves); thread: row=tid>>2, 16-elem segment
      const int row = tid >> 2, c0 = (tid & 3) * 16;
      _Float16* dst = As + row * 64 + c0;
      if (a_f16) {
        const uint4* src = (const uint4*)((const _Float16*)Ain +
                                          (size_t)(m0 + row) * D_MODEL + k0 + c0);
        *(uint4*)dst = src[0];
        *(uint4*)(dst + 8) = src[1];
      } else {
        const float4* src = (const float4*)((const float*)Ain +
                                            (size_t)(m0 + row) * D_MODEL + k0 + c0);
#pragma unroll
        for (int q = 0; q < 4; ++q) {
          float4 f = src[q];
          dst[q * 4 + 0] = (_Float16)f.x; dst[q * 4 + 1] = (_Float16)f.y;
          dst[q * 4 + 2] = (_Float16)f.z; dst[q * 4 + 3] = (_Float16)f.w;
        }
      }
    }
    {  // stage B^T tile (128x64 halves) from W rows (K-contiguous)
      const int row = tid >> 1, c0 = (tid & 1) * 32;
      _Float16* dst = Bs + row * 64 + c0;
      const float4* src = (const float4*)(W + (size_t)(n0 + row) * D_MODEL + k0 + c0);
#pragma unroll
      for (int q = 0; q < 8; ++q) {
        float4 f = src[q];
        dst[q * 4 + 0] = (_Float16)f.x; dst[q * 4 + 1] = (_Float16)f.y;
        dst[q * 4 + 2] = (_Float16)f.z; dst[q * 4 + 3] = (_Float16)f.w;
      }
    }
    __syncthreads();

#pragma unroll
    for (int ks = 0; ks < 64; ks += 32) {
      v16h a0 = frag_a(As + (wm * 32 + 0) * 64 + ks, 64);
      v16h a1 = frag_a(As + (wm * 32 + 16) * 64 + ks, 64);
      v16h b0 = frag_b(Bs + (wn * 32 + 0) * 64 + ks, 64);
      v16h b1 = frag_b(Bs + (wn * 32 + 16) * 64 + ks, 64);
      acc[0][0] = wmma16(a0, b0, acc[0][0]);
      acc[0][1] = wmma16(a0, b1, acc[0][1]);
      acc[1][0] = wmma16(a1, b0, acc[1][0]);
      acc[1][1] = wmma16(a1, b1, acc[1][1]);
    }
    __syncthreads();
  }

  // epilogue: C/D layout -> element (m = r + 8*(lane>>4), n = lane&15)
  const int lg = lane >> 4, ln = lane & 15;
#pragma unroll
  for (int mt = 0; mt < 2; ++mt)
#pragma unroll
    for (int nt = 0; nt < 2; ++nt)
#pragma unroll
      for (int r = 0; r < 8; ++r) {
        const int gm = m0 + wm * 32 + mt * 16 + r + lg * 8;
        const int gn = n0 + wn * 32 + nt * 16 + ln;
        const float v = acc[mt][nt][r] + bias[gn];
        if (mode == 0) {
          const int h = gn >> 6, dd = gn & 63;
          ((_Float16*)Out)[((size_t)h * SEQ + gm) * DK + dd] = (_Float16)v;
        } else if (mode == 1) {
          ((_Float16*)Out)[(size_t)gn * SEQ + gm] = (_Float16)v;  // V^T [h][d][s]
        } else {
          ((float*)Out)[(size_t)gm * D_MODEL + gn] = v;
        }
      }
}

// ---------------------------------------------------------------------------
// qrel[h][i][p] = dot(Q[h][i][:], rel_table[p][:])   (tiny; VALU is fine)
// ---------------------------------------------------------------------------
__global__ __launch_bounds__(256) void qrel_kernel(
    const _Float16* __restrict__ Qh, const float* __restrict__ rel,
    float* __restrict__ qrel)
{
  const int idx = blockIdx.x * 256 + threadIdx.x;
  if (idx >= NUM_HEADS * SEQ * NREL) return;
  const int p = idx % NREL;
  const int hi = idx / NREL;  // h*SEQ + i
  const _Float16* q = Qh + (size_t)hi * DK;
  const float* r = rel + (size_t)p * DK;
  float acc = 0.f;
#pragma unroll
  for (int d = 0; d < DK; ++d) acc += (float)q[d] * r[d];
  qrel[idx] = acc;
}

// ---------------------------------------------------------------------------
// Flash attention per (head, 64-row q tile). 8 waves; wave (wm = w&3 -> m-tile,
// wn = w>>2 -> pair of 16-wide n/d tiles). Online softmax with rel-pos bias.
// ---------------------------------------------------------------------------
__global__ __launch_bounds__(256) void attn_kernel(
    const _Float16* __restrict__ Qh, const _Float16* __restrict__ Kh,
    const _Float16* __restrict__ Vt, const float* __restrict__ qrel,
    const unsigned char* __restrict__ mask, _Float16* __restrict__ Xh)
{
  __shared__ __align__(32) _Float16 Qs[64 * 64];
  __shared__ __align__(32) _Float16 Ks[64 * 64];
  __shared__ __align__(32) _Float16 Vs[64 * 64];   // V^T chunk: [d][k]
  __shared__ __align__(32) _Float16 Pm[64 * 64];
  __shared__ float Ssm[64 * 64];
  __shared__ float qrel_s[64 * NREL];
  __shared__ float rowm[64], rowl[64], rowc[64];

  const int tid = threadIdx.x;
  const int q0 = blockIdx.x * 64;
  const int h = blockIdx.y;
  const int wave = tid >> 5, lane = tid & 31;
  const int wm = wave & 3, wn = wave >> 2;
  const int lg = lane >> 4, ln = lane & 15;

  {  // load Q tile once
    const int row = tid >> 2, c0 = (tid & 3) * 16;
    const uint4* src = (const uint4*)(Qh + ((size_t)h * SEQ + q0 + row) * DK + c0);
    uint4* dst = (uint4*)(Qs + row * 64 + c0);
    dst[0] = src[0]; dst[1] = src[1];
  }
  for (int idx = tid; idx < 64 * NREL; idx += 256) {
    const int row = idx / NREL, p = idx % NREL;
    qrel_s[idx] = qrel[((size_t)h * SEQ + q0 + row) * NREL + p];
  }
  if (tid < 64) { rowm[tid] = -1e30f; rowl[tid] = 0.f; rowc[tid] = 1.f; }
  __syncthreads();

  v8f oacc[2] = {};

  for (int j0 = 0; j0 < SEQ; j0 += 64) {
    {  // stage K chunk (row-major [k][d]) and V^T chunk (row-major [d][k])
      const int row = tid >> 2, c0 = (tid & 3) * 16;
      const uint4* ksrc = (const uint4*)(Kh + ((size_t)h * SEQ + j0 + row) * DK + c0);
      uint4* kdst = (uint4*)(Ks + row * 64 + c0);
      kdst[0] = ksrc[0]; kdst[1] = ksrc[1];
      const uint4* vsrc = (const uint4*)(Vt + ((size_t)h * DK + row) * SEQ + j0 + c0);
      uint4* vdst = (uint4*)(Vs + row * 64 + c0);
      vdst[0] = vsrc[0]; vdst[1] = vsrc[1];
    }
    __syncthreads();

    // S(64x64) = Q_tile @ K_chunk^T
#pragma unroll
    for (int nt = 0; nt < 2; ++nt) {
      const int ntile = wn * 2 + nt;
      v8f s_acc = {};
#pragma unroll
      for (int ks = 0; ks < 64; ks += 32) {
        v16h a = frag_a(Qs + wm * 16 * 64 + ks, 64);
        v16h b = frag_b(Ks + ntile * 16 * 64 + ks, 64);
        s_acc = wmma16(a, b, s_acc);
      }
#pragma unroll
      for (int r = 0; r < 8; ++r) {
        const int row = wm * 16 + r + lg * 8;
        const int col = ntile * 16 + ln;
        Ssm[row * 64 + col] = s_acc[r];
      }
    }
    __syncthreads();

    // online softmax: bias gather, scale 1/sqrt(64), mask, running max/sum
    if (tid < 64) {
      const int row = tid;
      const int i = q0 + row;
      const float mprev = rowm[row];
      float mx = mprev;
      for (int jj = 0; jj < 64; ++jj) {
        const int j = j0 + jj;
        int dpos = j - i;
        dpos = dpos < -PMAX ? -PMAX : (dpos > PMAX ? PMAX : dpos);
        float v = (Ssm[row * 64 + jj] + qrel_s[row * NREL + dpos + PMAX]) * 0.125f;
        if (!mask[(size_t)i * SEQ + j]) v = -1e12f;
        Ssm[row * 64 + jj] = v;
        mx = fmaxf(mx, v);
      }
      const float corr = __expf(mprev - mx);
      float sum = 0.f;
      for (int jj = 0; jj < 64; ++jj) {
        const float e = __expf(Ssm[row * 64 + jj] - mx);
        Pm[row * 64 + jj] = (_Float16)e;
        sum += e;
      }
      rowl[row] = rowl[row] * corr + sum;
      rowm[row] = mx;
      rowc[row] = corr;
    }
    __syncthreads();

    // rescale running O, then O += P @ V_chunk
#pragma unroll
    for (int nt = 0; nt < 2; ++nt)
#pragma unroll
      for (int r = 0; r < 8; ++r) {
        const int row = wm * 16 + r + lg * 8;
        oacc[nt][r] *= rowc[row];
      }
#pragma unroll
    for (int ks = 0; ks < 64; ks += 32) {
      v16h a = frag_a(Pm + wm * 16 * 64 + ks, 64);
#pragma unroll
      for (int nt = 0; nt < 2; ++nt) {
        v16h b = frag_b(Vs + (wn * 2 + nt) * 16 * 64 + ks, 64);
        oacc[nt] = wmma16(a, b, oacc[nt]);
      }
    }
    __syncthreads();
  }

  // finalize: O /= l, write f16 X[s][h*64+d] for the output projection
#pragma unroll
  for (int nt = 0; nt < 2; ++nt)
#pragma unroll
    for (int r = 0; r < 8; ++r) {
      const int row = wm * 16 + r + lg * 8;
      const float v = oacc[nt][r] / rowl[row];
      const int dd = (wn * 2 + nt) * 16 + ln;
      Xh[(size_t)(q0 + row) * D_MODEL + h * DK + dd] = (_Float16)v;
    }
}

// ---------------------------------------------------------------------------
// Host launch
// inputs: 0 query,1 key,2 value,3 mask,4 Wq,5 bq,6 Wk,7 bk,8 Wv,9 bv,
//         10 Wo,11 bo,12 rel_table
// ---------------------------------------------------------------------------
extern "C" void kernel_launch(void* const* d_in, const int* in_sizes, int n_in,
                              void* d_out, int out_size, void* d_ws, size_t ws_size,
                              hipStream_t stream) {
  (void)in_sizes; (void)n_in; (void)out_size; (void)ws_size;
  const float* query = (const float*)d_in[0];
  const float* key   = (const float*)d_in[1];
  const float* value = (const float*)d_in[2];
  const unsigned char* mask = (const unsigned char*)d_in[3];
  const float* Wq = (const float*)d_in[4];
  const float* bq = (const float*)d_in[5];
  const float* Wk = (const float*)d_in[6];
  const float* bk = (const float*)d_in[7];
  const float* Wv = (const float*)d_in[8];
  const float* bv = (const float*)d_in[9];
  const float* Wo = (const float*)d_in[10];
  const float* bo = (const float*)d_in[11];
  const float* rel = (const float*)d_in[12];

  // workspace layout (all 2-byte f16 counts; ~20.3 MB total)
  _Float16* Qh = (_Float16*)d_ws;                          // [16][2048][64]
  _Float16* Kh = Qh + (size_t)NUM_HEADS * SEQ * DK;        // [16][2048][64]
  _Float16* Vt = Kh + (size_t)NUM_HEADS * SEQ * DK;        // [16][64][2048]
  _Float16* Xh = Vt + (size_t)NUM_HEADS * SEQ * DK;        // [2048][1024]
  float*  qrel = (float*)(Xh + (size_t)SEQ * D_MODEL);     // [16][2048][33]

  dim3 ggrid(D_MODEL / 128, SEQ / 64);
  gemm_xwt_kernel<<<ggrid, 256, 0, stream>>>(query, 0, Wq, bq, Qh, 0);
  gemm_xwt_kernel<<<ggrid, 256, 0, stream>>>(key,   0, Wk, bk, Kh, 0);
  gemm_xwt_kernel<<<ggrid, 256, 0, stream>>>(value, 0, Wv, bv, Vt, 1);

  const int nq = NUM_HEADS * SEQ * NREL;
  qrel_kernel<<<(nq + 255) / 256, 256, 0, stream>>>(Qh, rel, qrel);

  attn_kernel<<<dim3(SEQ / 64, NUM_HEADS), 256, 0, stream>>>(Qh, Kh, Vt, qrel,
                                                             mask, Xh);

  gemm_xwt_kernel<<<ggrid, 256, 0, stream>>>(Xh, 1, Wo, bo, (float*)d_out, 2);
}
